// DABi_LSTM_28484223107589
// MI455X (gfx1250) — matrix operand order repaired and checked by reading, not compiled, not run on hardware
//
#include <hip/hip_runtime.h>

#define B_ 1024
#define T_ 100
#define M_ 128
#define H_ 128

typedef __attribute__((ext_vector_type(16))) _Float16 v16h;
typedef __attribute__((ext_vector_type(8)))  _Float16 v8h;
typedef __attribute__((ext_vector_type(8)))  float    v8f;

typedef unsigned int u32x4 __attribute__((ext_vector_type(4)));
typedef int          i32x4 __attribute__((ext_vector_type(4)));
typedef int          i32x8 __attribute__((ext_vector_type(8)));

#if __has_builtin(__builtin_amdgcn_tensor_load_to_lds) && \
    __has_builtin(__builtin_amdgcn_s_wait_tensorcnt)
#define HAVE_TDM 1
#else
#define HAVE_TDM 0
#endif

__device__ __forceinline__ v8f wmma16(v16h a, v16h b, v8f c) {
  // D = A(16x32) * B(32x16) + C, f32 accumulate
  return __builtin_amdgcn_wmma_f32_16x16x32_f16(false, a, false, b, (short)0, c, false, false);
}

// A-fragment (16-bit 16x32): lanes 0-15 hold M=lane, K={0..15}; lanes 16-31 K={16..31}
__device__ __forceinline__ v16h aFrag(const _Float16* rowp, int kbase, int half) {
  union { v16h v; v8h h[2]; } u;
  u.h[0] = *(const v8h*)(rowp + kbase + half * 16);
  u.h[1] = *(const v8h*)(rowp + kbase + half * 16 + 8);
  return u.v;
}

__device__ __forceinline__ float sigm(float x) { return 1.0f / (1.0f + __expf(-x)); }

#if HAVE_TDM
// TDM: async DMA of a 2D tile [16 rows x 128 f32], row stride T_*M_ elements, into LDS.
// D# group0: count=1, lds_addr, 57-bit global addr, type=2. group1: data_size=4B,
// tensor_dim0=128, tensor_dim1=16, tile_dim0=128, tile_dim1=16, dim0_stride=T_*M_.
__device__ __forceinline__ void tdm_load_tile16x128(unsigned lds_off, const float* gptr) {
  unsigned long long ga = (unsigned long long)(uintptr_t)gptr;
  u32x4 g0;
  g0[0] = 1u;                                         // count=1 valid descriptor
  g0[1] = lds_off;                                    // LDS byte address
  g0[2] = (unsigned)(ga & 0xFFFFFFFFu);               // global_addr[31:0]
  g0[3] = (unsigned)((ga >> 32) & 0x01FFFFFFu) | (2u << 30);  // addr[56:32] | type=2
  i32x8 g1;
  g1[0] = 0x00020000;                                 // data_size=2 (4 bytes)
  g1[1] = (int)(128u << 16);                          // tensor_dim0 = 128
  g1[2] = (int)(16u << 16);                           // tensor_dim1 = 16
  g1[3] = (int)(128u << 16);                          // tile_dim0 = 128
  g1[4] = 16;                                         // tile_dim1 = 16 (tile_dim2 = 0)
  g1[5] = (int)(T_ * M_);                             // tensor_dim0_stride
  g1[6] = 0;
  g1[7] = 0;
  i32x4 gz = {0, 0, 0, 0};
#if __clang_major__ >= 23
  i32x8 gz8 = {0, 0, 0, 0, 0, 0, 0, 0};
  __builtin_amdgcn_tensor_load_to_lds(g0, g1, gz, gz, gz8, 0);
#else
  __builtin_amdgcn_tensor_load_to_lds(g0, g1, gz, gz, 0);
#endif
}
#endif

// ---------------- alpha: softmax_m( X[b,:,m] . w_x )  (h,c terms cancel in softmax) --------
__global__ __launch_bounds__(128) void k_alpha(const float* __restrict__ X,
                                               const float* __restrict__ attnW,
                                               float* __restrict__ alpha) {
  __shared__ float red[128];
  const int b = blockIdx.x, m = threadIdx.x;
  const float* xp = X + (size_t)b * T_ * M_ + m;
  float s = 0.f;
  for (int t = 0; t < T_; ++t) s += xp[(size_t)t * M_] * attnW[2 * H_ + t];
  red[m] = s; __syncthreads();
  for (int o = 64; o > 0; o >>= 1) { if (m < o) red[m] = fmaxf(red[m], red[m + o]); __syncthreads(); }
  float mx = red[0]; __syncthreads();
  float e = __expf(s - mx);
  red[m] = e; __syncthreads();
  for (int o = 64; o > 0; o >>= 1) { if (m < o) red[m] += red[m + o]; __syncthreads(); }
  alpha[(size_t)b * M_ + m] = e / red[0];
}

// ---------------- weight repack to f16 ----------------
__global__ __launch_bounds__(256) void k_prep(
    const float* __restrict__ eWih, const float* __restrict__ eWhh,
    const float* __restrict__ ebih, const float* __restrict__ ebhh,
    const float* __restrict__ dW1,  const float* __restrict__ dWhh,
    const float* __restrict__ dbih, const float* __restrict__ dbhh,
    _Float16* __restrict__ WCE, _Float16* __restrict__ W1DC,
    _Float16* __restrict__ W1X, _Float16* __restrict__ WCD,
    float* __restrict__ BE, float* __restrict__ BD) {
  int i = blockIdx.x * 256 + threadIdx.x;
  if (i < 131072) { int n = i >> 8, k = i & 255;                      // [512][256] = [Wih | Whh]
    float v = (k < 128) ? eWih[n * 128 + k] : eWhh[n * 128 + (k - 128)];
    WCE[i] = (_Float16)v; return; }
  i -= 131072;
  if (i < 32768) { int h = i >> 8, k = i & 255;                       // [128][256] = W1[:, 0:2P]
    W1DC[i] = (_Float16)dW1[h * 384 + k]; return; }
  i -= 32768;
  if (i < 16384) { int h = i >> 7, k = i & 127;                       // [128][128] = W1[:, 2P:2P+H]
    W1X[i] = (_Float16)dW1[h * 384 + 256 + k]; return; }
  i -= 16384;
  if (i < 65536) { WCD[i] = (_Float16)dWhh[i]; return; }              // [512][128]
  i -= 65536;
  if (i < 512) { BE[i] = ebih[i] + ebhh[i]; return; }
  i -= 512;
  if (i < 512) { BD[i] = dbih[i] + dbhh[i]; return; }
}

// ---------------- encoder: 100 sequential LSTM steps, gates = [x~|h] @ [Wih;Whh]^T ----------
__global__ __launch_bounds__(256) void k_encoder(
    const float* __restrict__ X, const float* __restrict__ alpha,
    const _Float16* __restrict__ Wcat, const float* __restrict__ bias,
    float* __restrict__ Xenc, _Float16* __restrict__ Xenc16) {
  __shared__ _Float16 ldsA[16][256];   // cols 0..127 = x_tilde, 128..255 = h
#if HAVE_TDM
  __shared__ float xstage[2][16][128]; // TDM double buffer for X tiles
#endif
  const int tid = threadIdx.x;
  const int lane = tid & 31, wid = tid >> 5;
  const int half = lane >> 4, l16 = lane & 15;
  const int jcol = wid * 16 + l16;     // wave w owns gate columns {j, j+128, j+256, j+384}
  const int r0 = blockIdx.x * 16;
  const int xr = tid >> 4, xc = (tid & 15) * 8;

  float areg[8];
#pragma unroll
  for (int i = 0; i < 8; ++i) areg[i] = alpha[(r0 + xr) * M_ + xc + i];
  for (int idx = tid; idx < 16 * 128; idx += 256)
    ldsA[idx >> 7][128 + (idx & 127)] = (_Float16)0.f;

  const float bI = bias[jcol], bF = bias[128 + jcol], bG = bias[256 + jcol], bO = bias[384 + jcol];
  v8f cst = {0.f, 0.f, 0.f, 0.f, 0.f, 0.f, 0.f, 0.f};
  const v8f zf = {0.f, 0.f, 0.f, 0.f, 0.f, 0.f, 0.f, 0.f};

#if HAVE_TDM
  if (wid == 0)  // one wave issues the DMA; TENSORcnt is per-wave
    tdm_load_tile16x128((unsigned)(uintptr_t)&xstage[0][0][0],
                        X + (size_t)r0 * T_ * M_);
#endif

  for (int t = 0; t < T_; ++t) {
#if HAVE_TDM
    if (wid == 0) {
      if (t + 1 < T_) {
        // prefetch next step's tile, then wait for the current one (in-order per wave)
        tdm_load_tile16x128((unsigned)(uintptr_t)&xstage[(t + 1) & 1][0][0],
                            X + ((size_t)r0 * T_ + (t + 1)) * M_);
        __builtin_amdgcn_s_wait_tensorcnt(1);
      } else {
        __builtin_amdgcn_s_wait_tensorcnt(0);
      }
    }
    __syncthreads();  // publish TDM-written xstage[t&1] to all waves
    const float* xs = &xstage[t & 1][xr][xc];
#pragma unroll
    for (int i = 0; i < 8; ++i) ldsA[xr][xc + i] = (_Float16)(xs[i] * areg[i]);
#else
    const float* xp = X + ((size_t)(r0 + xr) * T_ + t) * M_ + xc;
#pragma unroll
    for (int i = 0; i < 8; ++i) ldsA[xr][xc + i] = (_Float16)(xp[i] * areg[i]);
#endif
    __syncthreads();

    v8f acc[4];
#pragma unroll
    for (int g = 0; g < 4; ++g) acc[g] = zf;
#pragma unroll
    for (int kc = 0; kc < 8; ++kc) {
      v16h a = aFrag(&ldsA[l16][0], kc * 32, half);
#pragma unroll
      for (int g = 0; g < 4; ++g) {
        v16h b = *(const v16h*)&Wcat[(size_t)(g * 128 + jcol) * 256 + kc * 32 + half * 16];
        acc[g] = wmma16(a, b, acc[g]);
      }
    }
    __syncthreads();

#pragma unroll
    for (int v = 0; v < 8; ++v) {
      const int row = v + half * 8;   // C-layout: VGPR v, lanes 0-15 -> M=v, lanes 16-31 -> M=v+8
      float gi = sigm(acc[0][v] + bI);
      float gf = sigm(acc[1][v] + bF);
      float gg = tanhf(acc[2][v] + bG);
      float go = sigm(acc[3][v] + bO);
      float c_ = gf * cst[v] + gi * gg;
      cst[v] = c_;
      float h_ = go * tanhf(c_);
      ldsA[row][128 + jcol] = (_Float16)h_;
      size_t gidx = ((size_t)(r0 + row) * T_ + t) * H_ + jcol;
      Xenc[gidx] = h_;
      Xenc16[gidx] = (_Float16)h_;
    }
  }
}

// ---------------- E = X_enc @ W1x^T + b1   (parallel GEMM over B*T rows) ------------------
__global__ __launch_bounds__(256) void k_eproj(
    const _Float16* __restrict__ Xenc16, const _Float16* __restrict__ W1x,
    const float* __restrict__ b1, float* __restrict__ E) {
  const int tid = threadIdx.x, lane = tid & 31, wid = tid >> 5;
  const int half = lane >> 4, l16 = lane & 15;
  const size_t r0 = (size_t)blockIdx.x * 16;
  const int ncol = wid * 16 + l16;
  v8f acc = {0.f, 0.f, 0.f, 0.f, 0.f, 0.f, 0.f, 0.f};
#pragma unroll
  for (int kc = 0; kc < 4; ++kc) {
    v16h a = aFrag(&Xenc16[(r0 + l16) * H_], kc * 32, half);
    v16h b = *(const v16h*)&W1x[(size_t)ncol * 128 + kc * 32 + half * 16];
    acc = wmma16(a, b, acc);
  }
  float bn = b1[ncol];
#pragma unroll
  for (int v = 0; v < 8; ++v) {
    size_t row = r0 + v + half * 8;
    E[row * H_ + ncol] = acc[v] + bn;
  }
}

// ---------------- decoder: 100 sequential steps (attn over T + LSTM) ----------------------
__global__ __launch_bounds__(256) void k_decoder(
    const float* __restrict__ Xenc, const float* __restrict__ E,
    const _Float16* __restrict__ W1dc, const float* __restrict__ w2,
    const _Float16* __restrict__ Whh, const float* __restrict__ biasD,
    const float* __restrict__ dWih, const float* __restrict__ fcW,
    const float* __restrict__ fcb, const float* __restrict__ fcfW,
    const float* __restrict__ fcfb, float* __restrict__ out) {
  __shared__ _Float16 ldsDC[16][256];  // cols 0..127 = d, 128..255 = cell c
  __shared__ float qctx[16][128];      // holds q, then reused for ctx
  __shared__ float sc[16][100];
  __shared__ float dbuf[16][128];
  __shared__ float w2s[128];
  __shared__ float yts[16];

  const int tid = threadIdx.x, lane = tid & 31, wid = tid >> 5;
  const int half = lane >> 4, l16 = lane & 15;
  const int jcol = wid * 16 + l16;
  const int r0 = blockIdx.x * 16;

  for (int idx = tid; idx < 16 * 256; idx += 256)
    ldsDC[idx >> 8][idx & 255] = (_Float16)0.f;
  if (tid < 128) w2s[tid] = w2[tid];

  float bD[4], wI[4];
#pragma unroll
  for (int g = 0; g < 4; ++g) { bD[g] = biasD[g * 128 + jcol]; wI[g] = dWih[g * 128 + jcol]; }
  v8f cst = {0.f, 0.f, 0.f, 0.f, 0.f, 0.f, 0.f, 0.f};
  const v8f zf = {0.f, 0.f, 0.f, 0.f, 0.f, 0.f, 0.f, 0.f};
  __syncthreads();

  for (int t = 0; t < T_; ++t) {
    // q = [d|c] @ W1dc^T   (wave w -> q cols 16w..16w+15)
    v8f qa = zf;
#pragma unroll
    for (int kc = 0; kc < 8; ++kc) {
      v16h a = aFrag(&ldsDC[l16][0], kc * 32, half);
      v16h b = *(const v16h*)&W1dc[(size_t)jcol * 256 + kc * 32 + half * 16];
      qa = wmma16(a, b, qa);
    }
#pragma unroll
    for (int v = 0; v < 8; ++v) qctx[v + half * 8][jcol] = qa[v];
    __syncthreads();

    // score[r][tq] = sum_h tanh(E + q) * w2   (E already has b1)
    for (int task = tid; task < 16 * T_; task += 256) {
      int r = task & 15, tq = task >> 4;
      const float* Ep = E + ((size_t)(r0 + r) * T_ + tq) * H_;
      if (task + 256 < 16 * T_) __builtin_prefetch(Ep + 16 * H_, 0, 1);  // next row (t+16)
      const float* qp = &qctx[r][0];
      float s = 0.f;
      for (int h = 0; h < H_; ++h) s += tanhf(Ep[h] + qp[h]) * w2s[h];
      sc[r][tq] = s;
    }
    __syncthreads();
    // softmax over T per row
    if (tid < 16) {
      float mx = -3.4e38f;
      for (int q = 0; q < T_; ++q) mx = fmaxf(mx, sc[tid][q]);
      float sum = 0.f;
      for (int q = 0; q < T_; ++q) { float e = __expf(sc[tid][q] - mx); sc[tid][q] = e; sum += e; }
      float inv = 1.f / sum;
      for (int q = 0; q < T_; ++q) sc[tid][q] *= inv;
    }
    __syncthreads();
    // ctx[r][h] = sum_t beta * X_enc
    for (int task = tid; task < 16 * H_; task += 256) {
      int r = task >> 7, h = task & 127;
      const float* Xp = Xenc + ((size_t)(r0 + r) * T_) * H_ + h;
      float s = 0.f;
      for (int q = 0; q < T_; ++q) s += sc[r][q] * Xp[(size_t)q * H_];
      qctx[r][h] = s;
    }
    __syncthreads();
    // y_tilde = ctx . fcW + fcb
    if (tid < 16) {
      float s = fcb[0];
      for (int h = 0; h < H_; ++h) s += qctx[tid][h] * fcW[h];
      yts[tid] = s;
    }
    __syncthreads();
    // gates = d @ Whh^T  (K=128)
    v8f acc[4];
#pragma unroll
    for (int g = 0; g < 4; ++g) acc[g] = zf;
#pragma unroll
    for (int kc = 0; kc < 4; ++kc) {
      v16h a = aFrag(&ldsDC[l16][0], kc * 32, half);
#pragma unroll
      for (int g = 0; g < 4; ++g) {
        v16h b = *(const v16h*)&Whh[(size_t)(g * 128 + jcol) * 128 + kc * 32 + half * 16];
        acc[g] = wmma16(a, b, acc[g]);
      }
    }
    __syncthreads();  // all ldsDC reads done before state overwrite
#pragma unroll
    for (int v = 0; v < 8; ++v) {
      const int row = v + half * 8;
      float yt = yts[row];
      float gi_ = sigm(acc[0][v] + bD[0] + yt * wI[0]);
      float gf_ = sigm(acc[1][v] + bD[1] + yt * wI[1]);
      float gg_ = tanhf(acc[2][v] + bD[2] + yt * wI[2]);
      float go_ = sigm(acc[3][v] + bD[3] + yt * wI[3]);
      float c_ = gf_ * cst[v] + gi_ * gg_;
      cst[v] = c_;
      float d_ = go_ * tanhf(c_);
      ldsDC[row][jcol] = (_Float16)d_;
      ldsDC[row][128 + jcol] = (_Float16)c_;
      if (t == T_ - 1) dbuf[row][jcol] = d_;
    }
    __syncthreads();
  }

  // out = [d | ctx] @ fcfW^T + fcfb
  for (int task = tid; task < 16 * 32; task += 256) {
    int r = task >> 5, cls = task & 31;
    float s = fcfb[cls];
    const float* wp = fcfW + (size_t)cls * 256;
    const float* dp = &dbuf[r][0];
    const float* cp = &qctx[r][0];
    for (int k = 0; k < H_; ++k) s += dp[k] * wp[k];
    for (int k = 0; k < H_; ++k) s += cp[k] * wp[128 + k];
    out[(size_t)(r0 + r) * 32 + cls] = s;
  }
}

extern "C" void kernel_launch(void* const* d_in, const int* in_sizes, int n_in,
                              void* d_out, int out_size, void* d_ws, size_t ws_size,
                              hipStream_t stream) {
  (void)in_sizes; (void)n_in; (void)out_size; (void)ws_size;
  const float* X     = (const float*)d_in[0];
  const float* eWih  = (const float*)d_in[1];
  const float* eWhh  = (const float*)d_in[2];
  const float* ebih  = (const float*)d_in[3];
  const float* ebhh  = (const float*)d_in[4];
  const float* eAttW = (const float*)d_in[5];
  const float* dW1   = (const float*)d_in[7];
  const float* db1   = (const float*)d_in[8];
  const float* dW2   = (const float*)d_in[9];
  const float* dWih  = (const float*)d_in[11];
  const float* dWhh  = (const float*)d_in[12];
  const float* dbih  = (const float*)d_in[13];
  const float* dbhh  = (const float*)d_in[14];
  const float* fcW   = (const float*)d_in[15];
  const float* fcb   = (const float*)d_in[16];
  const float* fcfW  = (const float*)d_in[17];
  const float* fcfb  = (const float*)d_in[18];

  char* ws = (char*)d_ws; size_t off = 0;
  auto take = [&](size_t bytes) -> void* {
    void* p = ws + off; off = (off + bytes + 255) & ~(size_t)255; return p;
  };
  _Float16* WCE    = (_Float16*)take((size_t)512 * 256 * 2);
  _Float16* W1DC   = (_Float16*)take((size_t)128 * 256 * 2);
  _Float16* W1X    = (_Float16*)take((size_t)128 * 128 * 2);
  _Float16* WCD    = (_Float16*)take((size_t)512 * 128 * 2);
  float*    BE     = (float*)take((size_t)512 * 4);
  float*    BD     = (float*)take((size_t)512 * 4);
  float*    ALPHA  = (float*)take((size_t)B_ * M_ * 4);
  float*    XENC   = (float*)take((size_t)B_ * T_ * H_ * 4);
  _Float16* XENC16 = (_Float16*)take((size_t)B_ * T_ * H_ * 2);
  float*    EB     = (float*)take((size_t)B_ * T_ * H_ * 4);

  k_alpha<<<B_, 128, 0, stream>>>(X, eAttW, ALPHA);
  k_prep<<<964, 256, 0, stream>>>(eWih, eWhh, ebih, ebhh, dW1, dWhh, dbih, dbhh,
                                  WCE, W1DC, W1X, WCD, BE, BD);
  k_encoder<<<B_ / 16, 256, 0, stream>>>(X, ALPHA, WCE, BE, XENC, XENC16);
  k_eproj<<<(B_ * T_) / 16, 256, 0, stream>>>(XENC16, W1X, db1, EB);
  k_decoder<<<B_ / 16, 256, 0, stream>>>(XENC, EB, W1DC, dW2, WCD, BD, dWih,
                                         fcW, fcb, fcfW, fcfb, (float*)d_out);
}
